// Codebook_33973191311621
// MI455X (gfx1250) — matrix operand order repaired and checked by reading, not compiled, not run on hardware
//
#include <hip/hip_runtime.h>

typedef __attribute__((ext_vector_type(16))) _Float16 v16h;
typedef __attribute__((ext_vector_type(8)))  float    v8f;

#define B_  8
#define T_  2048
#define D_  32
#define K_  8192
#define BT_ (B_ * T_)
#define EPSN 1e-12f

// ---------------------------------------------------------------------------
// Kernel A: invz[b][d] = 1 / max(sqrt(sum_t z[b,t,d]^2), eps)   (norm over T)
// ---------------------------------------------------------------------------
__global__ void colnorm_z_kernel(const float* __restrict__ z,
                                 float* __restrict__ invz) {
  __shared__ float part[256];
  const int b = blockIdx.x;          // 0..7
  const int d = threadIdx.x & 31;    // 0..31
  const int c = threadIdx.x >> 5;    // chunk 0..7
  const float* base = z + (size_t)b * T_ * D_;
  float s = 0.f;
  for (int t = c * 256; t < (c + 1) * 256; ++t) {
    float v = base[t * D_ + d];      // lanes read 128B contiguous
    s += v * v;
  }
  part[threadIdx.x] = s;
  __syncthreads();
  if (threadIdx.x < 32) {
    float acc = 0.f;
#pragma unroll
    for (int cc = 0; cc < 8; ++cc) acc += part[cc * 32 + d];
    invz[b * 32 + d] = 1.0f / fmaxf(sqrtf(acc), EPSN);
  }
}

// ---------------------------------------------------------------------------
// Kernel B: esq[k] = |e_k|^2 ; eh[k][d] = f16(e[k][d] / max(|e_k|, eps))
// Row-major f16 codebook is exactly the WMMA B-matrix feed (see kernel C).
// ---------------------------------------------------------------------------
__global__ void prep_codebook_kernel(const float* __restrict__ emb,
                                     _Float16* __restrict__ eh,
                                     float* __restrict__ esq) {
  const int lane = threadIdx.x & 31;
  const int wave = threadIdx.x >> 5;          // 0..7
  const int k0 = blockIdx.x * 64 + wave * 8;  // 8 rows per wave
  for (int i = 0; i < 8; ++i) {
    const int k = k0 + i;
    float v = emb[(size_t)k * 32 + lane];     // coalesced 128B per wave
    float s = v * v;
#pragma unroll
    for (int m = 16; m >= 1; m >>= 1) s += __shfl_xor(s, m, 32);
    float inv = 1.0f / fmaxf(sqrtf(s), EPSN);
    eh[(size_t)k * 32 + lane] = (_Float16)(v * inv);
    if (lane == 0) esq[k] = s;
  }
}

// ---------------------------------------------------------------------------
// Kernel C: fused distance GEMM + argmin.
// One wave owns a 16-row M-tile; D=32 -> exactly ONE v_wmma_f32_16x16x32_f16
// per 16x16 distance tile.  A = f16(-2 * zn rows) staged via LDS into the ISA
// A-layout; B = f16 normalized codebook streamed from L2 (16 halfs/lane =
// one column of K{0..7,16..23} or K{8..15,24..31}, i.e. two b128 loads).  Running min+idx
// kept in the WMMA C-layout, reduced with xor-shuffles at the end.
// Tile loop unrolled 4x so four independent WMMA result streams hide the
// WMMA->VALU RAW hazard (4 co-exec slots for F16 WMMA) without v_nops.
// ---------------------------------------------------------------------------
__global__ void __launch_bounds__(256, 2)
vq_argmin_kernel(const float* __restrict__ z,
                 const float* __restrict__ invz,
                 const _Float16* __restrict__ eh,
                 const float* __restrict__ esq,
                 int* __restrict__ idx_out) {
  __shared__ _Float16 sh[8 * 512];            // 16 rows x 32 halfs per wave
  const int lane = threadIdx.x & 31;
  const int wave = threadIdx.x >> 5;
  const int m0 = (blockIdx.x * 8 + wave) * 16;
  const int b = m0 / T_;                      // 16 rows never straddle b

  // Stage A rows: fold (-2 * invz) into the f32->f16 conversion.
  const float iz = invz[b * 32 + lane] * -2.0f;
  _Float16* wsh = sh + wave * 512;
  for (int i = 0; i < 16; ++i) {
    float v = z[(size_t)(m0 + i) * 32 + lane] * iz;
    wsh[i * 32 + lane] = (_Float16)v;
  }
  __syncthreads();

  // Build A v16h per ISA layout: lo lanes hold K{0..7,16..23}, hi lanes
  // K{8..15,24..31} of row M = lane&15.
  const int row = lane & 15;
  const int hi  = lane >> 4;
  union { uint4 u[2]; v16h h; } ua;
  const _Float16* abase = wsh + row * 32 + hi * 8;
  ua.u[0] = *(const uint4*)(abase);
  ua.u[1] = *(const uint4*)(abase + 16);
  const v16h A = ua.h;

  float    minv[8];
  unsigned mini[8];
#pragma unroll
  for (int j = 0; j < 8; ++j) { minv[j] = 3.4e38f; mini[j] = 0u; }

  const int nsub = lane & 15;                 // this lane's N within a tile
  const _Float16* bptr = eh + (size_t)nsub * 32 + hi * 8;
  const float*    eptr = esq + nsub;
  const v8f czero = {};

#pragma unroll 4
  for (int tile = 0; tile < K_ / 16; ++tile) {
    union { uint4 u[2]; v16h h; } ub;
    ub.u[0] = *(const uint4*)(bptr);          // column n, K{0..7,16..23} / K{8..15,24..31}
    ub.u[1] = *(const uint4*)(bptr + 16);
    __builtin_prefetch(bptr + 8 * 512, 0, 3); // WGP-scope: pull into near cache

    // acc = A x B  ( = -2 * zn . e_norm^T, f32 accumulate)
    v8f acc = __builtin_amdgcn_wmma_f32_16x16x32_f16(
        false, A, false, ub.h, (short)0, czero, false, false);

    const float es = *eptr;                   // |e_n|^2 (same for all 8 M)
    const unsigned n = (unsigned)(tile * 16 + nsub);
#pragma unroll
    for (int j = 0; j < 8; ++j) {
      float dv = acc[j] + es;
      bool lt = dv < minv[j];                 // strict < keeps first min
      minv[j] = lt ? dv : minv[j];
      mini[j] = lt ? n  : mini[j];
    }
    bptr += 16 * 32;                          // next 16 codebook rows (1 KB)
    eptr += 16;
  }

  // Argmin across the 16 lanes of each half-group (xor masks stay in-half).
#pragma unroll
  for (int m = 1; m <= 8; m <<= 1) {
#pragma unroll
    for (int j = 0; j < 8; ++j) {
      float    ov = __shfl_xor(minv[j], m, 32);
      unsigned oi = __shfl_xor(mini[j], m, 32);
      bool take = (ov < minv[j]) || (ov == minv[j] && oi < mini[j]);
      minv[j] = take ? ov : minv[j];
      mini[j] = take ? oi : mini[j];
    }
  }
  if (nsub == 0) {                            // lanes 0 (M 0..7) and 16 (M 8..15)
#pragma unroll
    for (int j = 0; j < 8; ++j)
      idx_out[m0 + hi * 8 + j] = (int)mini[j];
  }
}

// ---------------------------------------------------------------------------
// Kernel D: invq[b][d] = 1 / max(sqrt(sum_t e[idx[b,t]][d]^2), eps)
// ---------------------------------------------------------------------------
__global__ void colnorm_q_kernel(const float* __restrict__ emb,
                                 const int* __restrict__ idx,
                                 float* __restrict__ invq) {
  __shared__ float part[256];
  const int b = blockIdx.x;
  const int d = threadIdx.x & 31;
  const int c = threadIdx.x >> 5;
  float s = 0.f;
  for (int t = c * 256; t < (c + 1) * 256; ++t) {
    int k = idx[b * T_ + t];
    float v = emb[(size_t)k * 32 + d];
    s += v * v;
  }
  part[threadIdx.x] = s;
  __syncthreads();
  if (threadIdx.x < 32) {
    float acc = 0.f;
#pragma unroll
    for (int cc = 0; cc < 8; ++cc) acc += part[cc * 32 + d];
    invq[b * 32 + d] = 1.0f / fmaxf(sqrtf(acc), EPSN);
  }
}

// ---------------------------------------------------------------------------
// Kernel E: z_q_st (== normalized gathered embedding) + per-block SSE partials.
// ---------------------------------------------------------------------------
__global__ void finalize_kernel(const float* __restrict__ z,
                                const float* __restrict__ emb,
                                const int* __restrict__ idx,
                                const float* __restrict__ invz,
                                const float* __restrict__ invq,
                                float* __restrict__ zq_out,
                                float* __restrict__ partials) {
  __shared__ float red[256];
  const int gid = blockIdx.x * 256 + threadIdx.x;
  const int m = gid >> 5;
  const int d = gid & 31;
  const int b = m / T_;
  const int k = idx[m];
  float zq = emb[(size_t)k * 32 + d] * invq[b * 32 + d];
  float zn = z[(size_t)gid] * invz[b * 32 + d];
  zq_out[gid] = zq;                           // z_q_st == z_q numerically
  float diff = zq - zn;
  red[threadIdx.x] = diff * diff;
  __syncthreads();
  for (int s2 = 128; s2 > 0; s2 >>= 1) {
    if (threadIdx.x < s2) red[threadIdx.x] += red[threadIdx.x + s2];
    __syncthreads();
  }
  if (threadIdx.x == 0) partials[blockIdx.x] = red[0];
}

// ---------------------------------------------------------------------------
// Kernel F: deterministic final loss reduction: 1.25 * mean(diff^2)
// ---------------------------------------------------------------------------
__global__ void loss_kernel(const float* __restrict__ partials,
                            float* __restrict__ loss) {
  __shared__ float red[256];
  float s = 0.f;
  for (int i = threadIdx.x; i < 2048; i += 256) s += partials[i];
  red[threadIdx.x] = s;
  __syncthreads();
  for (int s2 = 128; s2 > 0; s2 >>= 1) {
    if (threadIdx.x < s2) red[threadIdx.x] += red[threadIdx.x + s2];
    __syncthreads();
  }
  if (threadIdx.x == 0)
    loss[0] = red[0] * (1.25f / (float)(BT_ * D_));
}

// ---------------------------------------------------------------------------
extern "C" void kernel_launch(void* const* d_in, const int* in_sizes, int n_in,
                              void* d_out, int out_size, void* d_ws, size_t ws_size,
                              hipStream_t stream) {
  const float* z   = (const float*)d_in[0];   // [8,2048,32]
  const float* emb = (const float*)d_in[1];   // [8192,32]

  float* out    = (float*)d_out;
  float* zq_out = out;                        // 524288 floats
  int*   idx_o  = (int*)(out + (size_t)BT_ * D_);   // 16384 int32
  float* loss_o = out + (size_t)BT_ * D_ + BT_;     // 1 float

  char* w = (char*)d_ws;
  _Float16* eh     = (_Float16*)(w);              // 512 KB
  float*    esq    = (float*)(w + 524288);        // 32 KB
  float*    invz   = (float*)(w + 557056);        // 1 KB
  float*    invq   = (float*)(w + 558080);        // 1 KB
  float*    parts  = (float*)(w + 559104);        // 8 KB

  colnorm_z_kernel<<<B_, 256, 0, stream>>>(z, invz);
  prep_codebook_kernel<<<K_ / 64, 256, 0, stream>>>(emb, eh, esq);
  vq_argmin_kernel<<<BT_ / 16 / 8, 256, 0, stream>>>(z, invz, eh, esq, idx_o);
  colnorm_q_kernel<<<B_, 256, 0, stream>>>(emb, idx_o, invq);
  finalize_kernel<<<(BT_ * D_) / 256, 256, 0, stream>>>(z, emb, idx_o, invz,
                                                        invq, zq_out, parts);
  loss_kernel<<<1, 256, 0, stream>>>(parts, loss_o);
}
